// SubGraph_54769422958874
// MI455X (gfx1250) — matrix operand order repaired and checked by reading, not compiled, not run on hardware
//
#include <hip/hip_runtime.h>

typedef __attribute__((ext_vector_type(16))) _Float16 v16h;
typedef __attribute__((ext_vector_type(4)))  _Float16 v4h;
typedef __attribute__((ext_vector_type(8)))  float    v8f;

#define B_     256
#define N_     512
#define H_     128
#define NH_    2
#define DH_    64
#define DEPTH_ 3
#define R_     (B_*N_)    /* 131072 rows */
#define MT_    (R_/16)    /* 8192 16-row tiles */
#define WSZ    (H_*H_)    /* 16384 elements per weight matrix */

__device__ __forceinline__ v8f wmma_f16(const v16h& a, const v16h& b, v8f c) {
  return __builtin_amdgcn_wmma_f32_16x16x32_f16(false, a, false, b, (short)0, c, false, false);
}

// -------- one-shot: convert f32 weights -> f16 pre-swizzled into WMMA B-fragment order ----
// Packed layout per matrix: [ct(8)][kc(4)][lane(32)][j(16)] halves, where
//   element = W[(kc*32 + (lane>>4)*16 + j) * H + ct*16 + (lane&15)]
__global__ void pack_weights(const float* __restrict__ W0, const float* __restrict__ W1,
                             const float* __restrict__ Wq, const float* __restrict__ Wk,
                             const float* __restrict__ Wv, _Float16* __restrict__ dst) {
  int i = blockIdx.x * 256 + threadIdx.x;      // total 11*WSZ
  int mat = i >> 14;                            // /16384
  int r   = i & (WSZ - 1);
  int j    = r & 15;
  int lane = (r >> 4) & 31;
  int kc   = (r >> 9) & 3;
  int ct   = (r >> 11) & 7;
  int k = kc * 32 + (lane >> 4) * 16 + j;
  int n = ct * 16 + (lane & 15);
  const float* src;
  if      (mat <  1) src = W0 + (size_t)(mat - 0) * WSZ;
  else if (mat <  2) src = W1 + (size_t)(mat - 1) * WSZ;
  else if (mat <  5) src = Wq + (size_t)(mat - 2) * WSZ;
  else if (mat <  8) src = Wk + (size_t)(mat - 5) * WSZ;
  else               src = Wv + (size_t)(mat - 8) * WSZ;
  dst[i] = (_Float16)src[(size_t)k * H_ + n];
}

// -------- fused 16-row GEMM(H x H, packed B) + bias + LayerNorm + ReLU tile step --------
__device__ __forceinline__ void tile_gemm_ln_relu(
    _Float16 (*sX)[H_], float (*sY)[H_], float (*sRed)[32], float (*sMV)[2],
    const _Float16* __restrict__ Wpk, const float* __restrict__ bias,
    const float* __restrict__ gamma, const float* __restrict__ beta, int t) {
  const int wave = t >> 5, lane = t & 31, m = lane & 15, hi = lane >> 4;
  const v16h* Wp = (const v16h*)Wpk;
  v8f c = {};
#pragma unroll
  for (int kc = 0; kc < 4; ++kc) {               // K = 128 in chunks of 32
    v16h a;
    int kbA = kc * 32 + hi * 16;
#pragma unroll
    for (int j = 0; j < 8; ++j) { a[j] = sX[m][kbA + j]; a[j + 8] = sX[m][kbA + 8 + j]; }
    v16h bv = Wp[(size_t)(wave * 4 + kc) * 32 + lane];
    c = wmma_f16(a, bv, c);
  }
#pragma unroll
  for (int r = 0; r < 8; ++r) {
    int mm = r + hi * 8, cc = wave * 16 + m;
    sY[mm][cc] = c[r] + bias[cc];
  }
  __syncthreads();
  // LayerNorm + ReLU: 16 threads per row, two contiguous float4 groups each
  int r = t >> 4, l = t & 15;
  float4* yrow = (float4*)sY[r];
  float4 u0 = yrow[l], u1 = yrow[l + 16];
  float s1 = u0.x + u0.y + u0.z + u0.w + u1.x + u1.y + u1.z + u1.w;
  float s2 = u0.x*u0.x + u0.y*u0.y + u0.z*u0.z + u0.w*u0.w
           + u1.x*u1.x + u1.y*u1.y + u1.z*u1.z + u1.w*u1.w;
  sRed[r][l] = s1; sRed[r][l + 16] = s2;
  __syncthreads();
  if (l == 0) {
    float a1 = 0.f, a2 = 0.f;
#pragma unroll
    for (int i = 0; i < 16; ++i) { a1 += sRed[r][i]; a2 += sRed[r][16 + i]; }
    float mean = a1 * (1.f / H_);
    float var  = a2 * (1.f / H_) - mean * mean;
    sMV[r][0] = mean; sMV[r][1] = rsqrtf(var + 1e-5f);
  }
  __syncthreads();
  float mean = sMV[r][0], rstd = sMV[r][1];
  const float4* g4 = (const float4*)gamma;
  const float4* q4 = (const float4*)beta;
  float4 g0 = g4[l], g1 = g4[l + 16], p0 = q4[l], p1 = q4[l + 16];
  float4 o0, o1;
  o0.x = fmaxf((u0.x - mean) * rstd * g0.x + p0.x, 0.f);
  o0.y = fmaxf((u0.y - mean) * rstd * g0.y + p0.y, 0.f);
  o0.z = fmaxf((u0.z - mean) * rstd * g0.z + p0.z, 0.f);
  o0.w = fmaxf((u0.w - mean) * rstd * g0.w + p0.w, 0.f);
  o1.x = fmaxf((u1.x - mean) * rstd * g1.x + p1.x, 0.f);
  o1.y = fmaxf((u1.y - mean) * rstd * g1.y + p1.y, 0.f);
  o1.z = fmaxf((u1.z - mean) * rstd * g1.z + p1.z, 0.f);
  o1.w = fmaxf((u1.w - mean) * rstd * g1.w + p1.w, 0.f);
  yrow[l] = o0; yrow[l + 16] = o1;
  v4h h0 = { (_Float16)o0.x, (_Float16)o0.y, (_Float16)o0.z, (_Float16)o0.w };
  v4h h1 = { (_Float16)o1.x, (_Float16)o1.y, (_Float16)o1.z, (_Float16)o1.w };
  *(v4h*)&sX[r][4 * l]      = h0;
  *(v4h*)&sX[r][64 + 4 * l] = h1;
  __syncthreads();
}

// -------- MLP: two chained Linear+LN+ReLU layers, 16 rows per block --------
__global__ __launch_bounds__(256) void mlp_kernel(
    const float* __restrict__ X, const _Float16* __restrict__ Wh,
    const float* __restrict__ b0, const float* __restrict__ g0, const float* __restrict__ be0,
    const float* __restrict__ b1, const float* __restrict__ g1, const float* __restrict__ be1,
    float* __restrict__ Hf32, _Float16* __restrict__ Hf16) {
  __shared__ _Float16 sX[16][H_];
  __shared__ float    sY[16][H_];
  __shared__ float    sRed[16][32];
  __shared__ float    sMV[16][2];
  int t = threadIdx.x;
  size_t row0 = (size_t)blockIdx.x * 16;
  {
    const float4* xs = (const float4*)(X + row0 * H_);
#pragma unroll
    for (int i = t; i < 512; i += 256) {
      float4 v4 = xs[i];
      int r = i >> 5, c4 = (i & 31) * 4;
      v4h h = { (_Float16)v4.x, (_Float16)v4.y, (_Float16)v4.z, (_Float16)v4.w };
      *(v4h*)&sX[r][c4] = h;
    }
  }
  __syncthreads();
  tile_gemm_ln_relu(sX, sY, sRed, sMV, Wh,       b0, g0, be0, t);
  tile_gemm_ln_relu(sX, sY, sRed, sMV, Wh + WSZ, b1, g1, be1, t);
  {
    float4* hd = (float4*)(Hf32 + row0 * H_);
    const float4* ys = (const float4*)sY;
#pragma unroll
    for (int i = t; i < 512; i += 256) hd[i] = ys[i];
    uint4* hh = (uint4*)(Hf16 + row0 * H_);
    const uint4* xs16 = (const uint4*)sX;
    hh[t] = xs16[t];                        // 256 uint4 = 16x128 halves
  }
}

// -------- q/k/v projections: grid.y selects matrix; V stored transposed --------
__global__ __launch_bounds__(256) void proj_kernel(
    const _Float16* __restrict__ Hf16,
    const _Float16* __restrict__ Wq16, const _Float16* __restrict__ Wk16, const _Float16* __restrict__ Wv16,
    const float* __restrict__ bq, const float* __restrict__ bk, const float* __restrict__ bv,
    _Float16* __restrict__ Qo, _Float16* __restrict__ Ko, _Float16* __restrict__ Vt) {
  __shared__ _Float16 sX[16][H_];
  int t = threadIdx.x;
  const int wave = t >> 5, lane = t & 31, m = lane & 15, hi = lane >> 4;
  size_t row0 = (size_t)blockIdx.x * 16;
  const _Float16* W   = blockIdx.y == 0 ? Wq16 : (blockIdx.y == 1 ? Wk16 : Wv16);
  const float*   bias = blockIdx.y == 0 ? bq   : (blockIdx.y == 1 ? bk   : bv);
  {
    const uint4* src = (const uint4*)(Hf16 + row0 * H_);
    uint4* dst = (uint4*)sX;
    dst[t] = src[t];
  }
  __syncthreads();
  const v16h* Wp = (const v16h*)W;
  v8f c = {};
#pragma unroll
  for (int kc = 0; kc < 4; ++kc) {
    v16h a;
    int kbA = kc * 32 + hi * 16;
#pragma unroll
    for (int j = 0; j < 8; ++j) { a[j] = sX[m][kbA + j]; a[j + 8] = sX[m][kbA + 8 + j]; }
    v16h bv2 = Wp[(size_t)(wave * 4 + kc) * 32 + lane];
    c = wmma_f16(a, bv2, c);
  }
  const int cc = wave * 16 + m;
  if (blockIdx.y == 2) {
    // transposed V: Vt[b][head][dh][n], contiguous along n (key index)
    int brow  = (int)(row0 >> 9);        // batch (16-row tiles never straddle a batch)
    int nbase = (int)(row0 & 511);
    int head = cc >> 6, dh = cc & 63;
    _Float16* vcol = Vt + ((size_t)brow * NH_ + head) * DH_ * N_ + (size_t)dh * N_ + nbase;
    float bc = bias[cc];
#pragma unroll
    for (int r = 0; r < 8; ++r) vcol[r + hi * 8] = (_Float16)(c[r] + bc);
  } else {
    _Float16* Out = blockIdx.y == 0 ? Qo : Ko;
    float bc = bias[cc];
#pragma unroll
    for (int r = 0; r < 8; ++r) {
      int mm = r + hi * 8;
      Out[(row0 + mm) * H_ + cc] = (_Float16)(c[r] + bc);
    }
  }
}

// -------- masked 2-head attention: block = (batch, head, 16-query tile) --------
__global__ __launch_bounds__(256) void attn_kernel(
    const _Float16* __restrict__ Q, const _Float16* __restrict__ K,
    const _Float16* __restrict__ Vt, const int* __restrict__ lengths,
    float* __restrict__ Ctx) {
  __shared__ float    sS[16][N_];     // 32 KB scores
  __shared__ _Float16 sP[16][N_];     // 16 KB probs
  __shared__ float    sRed[16][32];
  __shared__ float    sMV[16][2];
  const int t = threadIdx.x;
  const int wave = t >> 5, lane = t & 31, m = lane & 15, hi = lane >> 4;
  const int b = blockIdx.z, head = blockIdx.y, n0 = blockIdx.x * 16;
  const int len = lengths[b];
  const size_t rowbase = (size_t)b * N_;
  const _Float16* qb = Q + rowbase * H_ + head * DH_;
  const _Float16* kb = K + rowbase * H_ + head * DH_;
  const _Float16* vtb = Vt + ((size_t)b * NH_ + head) * DH_ * N_;   // [64][512]

  // q tile A-fragments (K = 64 -> two chunks of 32), per wave
  v16h a0, a1;
  {
    const _Float16* qr = qb + (size_t)(n0 + m) * H_;
    int kbA = hi * 16;
#pragma unroll
    for (int j = 0; j < 8; ++j) {
      a0[j] = qr[kbA + j];      a0[j + 8] = qr[kbA + 8 + j];
      a1[j] = qr[32 + kbA + j]; a1[j + 8] = qr[32 + kbA + 8 + j];
    }
  }
  // scores: each wave covers 4 key tiles of 16
#pragma unroll
  for (int kt0 = 0; kt0 < 4; ++kt0) {
    int kt = wave * 4 + kt0;
    v16h b0v, b1v;
    const _Float16* kr = kb + (size_t)(kt * 16 + m) * H_;
    int kbB = hi * 16;
#pragma unroll
    for (int j = 0; j < 16; ++j) { b0v[j] = kr[kbB + j]; b1v[j] = kr[32 + kbB + j]; }
    v8f c = {};
    c = wmma_f16(a0, b0v, c);
    c = wmma_f16(a1, b1v, c);
#pragma unroll
    for (int r = 0; r < 8; ++r) {
      int mm = r + hi * 8;
      int row = n0 + mm, kk = kt * 16 + m;
      float biasv = (row < len && kk < len) ? 0.0f : -10000.0f;
      sS[mm][kk] = c[r] * 0.125f + biasv;   // 1/sqrt(64)
    }
  }
  __syncthreads();
  // row softmax over 512: register-resident, float4-vectorized
  {
    int r = t >> 4, l = t & 15;
    const float4* srow = (const float4*)sS[r];
    float4 vv[8];
    float mx = -3.0e38f;
#pragma unroll
    for (int g = 0; g < 8; ++g) {
      vv[g] = srow[l + 16 * g];
      mx = fmaxf(mx, fmaxf(fmaxf(vv[g].x, vv[g].y), fmaxf(vv[g].z, vv[g].w)));
    }
    sRed[r][l] = mx;
    __syncthreads();
    if (l == 0) { float v = sRed[r][0]; for (int i = 1; i < 16; ++i) v = fmaxf(v, sRed[r][i]); sMV[r][0] = v; }
    __syncthreads();
    float rowmax = sMV[r][0], sum = 0.f;
#pragma unroll
    for (int g = 0; g < 8; ++g) {
      vv[g].x = __expf(vv[g].x - rowmax); vv[g].y = __expf(vv[g].y - rowmax);
      vv[g].z = __expf(vv[g].z - rowmax); vv[g].w = __expf(vv[g].w - rowmax);
      sum += vv[g].x + vv[g].y + vv[g].z + vv[g].w;
    }
    sRed[r][l] = sum;
    __syncthreads();
    if (l == 0) { float v = 0.f; for (int i = 0; i < 16; ++i) v += sRed[r][i]; sMV[r][1] = 1.0f / v; }
    __syncthreads();
    float inv = sMV[r][1];
#pragma unroll
    for (int g = 0; g < 8; ++g) {
      v4h h = { (_Float16)(vv[g].x * inv), (_Float16)(vv[g].y * inv),
                (_Float16)(vv[g].z * inv), (_Float16)(vv[g].w * inv) };
      *(v4h*)&sP[r][(l + 16 * g) * 4] = h;
    }
  }
  __syncthreads();
  // ctx = P @ V : waves 0..3 each own a 16-col slice of DH=64; K = 512 -> 16 WMMAs.
  if (wave < 4) {
    const _Float16* vcol = vtb + (size_t)(wave * 16 + m) * N_;
    v8f c = {};
#pragma unroll
    for (int kc = 0; kc < 16; ++kc) {
      v16h a;
      int kbA = kc * 32 + hi * 16;
#pragma unroll
      for (int j = 0; j < 8; ++j) { a[j] = sP[m][kbA + j]; a[j + 8] = sP[m][kbA + 8 + j]; }
      v16h bv2 = *(const v16h*)(vcol + kc * 32 + hi * 16);
      c = wmma_f16(a, bv2, c);
    }
#pragma unroll
    for (int r = 0; r < 8; ++r) {
      int mm = r + hi * 8;
      Ctx[(rowbase + n0 + mm) * H_ + head * DH_ + wave * 16 + m] = c[r];
    }
  }
}

// -------- relu(ctx) + residual + LayerNorm, in-place h update (register-resident) --------
__global__ __launch_bounds__(256) void ln_res_kernel(
    const float* __restrict__ Ctx, float* __restrict__ Hf32, _Float16* __restrict__ Hf16,
    const float* __restrict__ gamma, const float* __restrict__ beta) {
  __shared__ float sRed[16][32];
  __shared__ float sMV[16][2];
  int t = threadIdx.x, r = t >> 4, l = t & 15;
  size_t row = (size_t)blockIdx.x * 16 + r;
  const float4* c4p = (const float4*)(Ctx  + row * H_);
  float4*       h4p = (float4*)      (Hf32 + row * H_);
  float4 c0 = c4p[l], c1 = c4p[l + 16];
  float4 h0 = h4p[l], h1 = h4p[l + 16];
  float4 u0, u1;
  u0.x = fmaxf(c0.x, 0.f) + h0.x; u0.y = fmaxf(c0.y, 0.f) + h0.y;
  u0.z = fmaxf(c0.z, 0.f) + h0.z; u0.w = fmaxf(c0.w, 0.f) + h0.w;
  u1.x = fmaxf(c1.x, 0.f) + h1.x; u1.y = fmaxf(c1.y, 0.f) + h1.y;
  u1.z = fmaxf(c1.z, 0.f) + h1.z; u1.w = fmaxf(c1.w, 0.f) + h1.w;
  float s1 = u0.x + u0.y + u0.z + u0.w + u1.x + u1.y + u1.z + u1.w;
  float s2 = u0.x*u0.x + u0.y*u0.y + u0.z*u0.z + u0.w*u0.w
           + u1.x*u1.x + u1.y*u1.y + u1.z*u1.z + u1.w*u1.w;
  sRed[r][l] = s1; sRed[r][l + 16] = s2;
  __syncthreads();
  if (l == 0) {
    float a1 = 0.f, a2 = 0.f;
#pragma unroll
    for (int i = 0; i < 16; ++i) { a1 += sRed[r][i]; a2 += sRed[r][16 + i]; }
    float mean = a1 * (1.f / H_);
    float var  = a2 * (1.f / H_) - mean * mean;
    sMV[r][0] = mean; sMV[r][1] = rsqrtf(var + 1e-5f);
  }
  __syncthreads();
  float mean = sMV[r][0], rstd = sMV[r][1];
  const float4* g4 = (const float4*)gamma;
  const float4* q4 = (const float4*)beta;
  float4 g0 = g4[l], g1 = g4[l + 16], p0 = q4[l], p1 = q4[l + 16];
  float4 o0, o1;
  o0.x = (u0.x - mean) * rstd * g0.x + p0.x; o0.y = (u0.y - mean) * rstd * g0.y + p0.y;
  o0.z = (u0.z - mean) * rstd * g0.z + p0.z; o0.w = (u0.w - mean) * rstd * g0.w + p0.w;
  o1.x = (u1.x - mean) * rstd * g1.x + p1.x; o1.y = (u1.y - mean) * rstd * g1.y + p1.y;
  o1.z = (u1.z - mean) * rstd * g1.z + p1.z; o1.w = (u1.w - mean) * rstd * g1.w + p1.w;
  h4p[l] = o0; h4p[l + 16] = o1;
  v4h e0 = { (_Float16)o0.x, (_Float16)o0.y, (_Float16)o0.z, (_Float16)o0.w };
  v4h e1 = { (_Float16)o1.x, (_Float16)o1.y, (_Float16)o1.z, (_Float16)o1.w };
  _Float16* hh = Hf16 + row * H_;
  *(v4h*)&hh[4 * l]      = e0;
  *(v4h*)&hh[64 + 4 * l] = e1;
}

// -------- final max over N: 4 row-partitions x 32 lanes of float4 --------
__global__ __launch_bounds__(128) void max_kernel(const float* __restrict__ Hf32,
                                                  float* __restrict__ Out) {
  __shared__ float4 sm[4][32];
  int t = threadIdx.x, c4 = t & 31, rq = t >> 5;
  int b = blockIdx.x;
  const float4* p = (const float4*)(Hf32 + (size_t)b * N_ * H_);
  float4 mx = { -3.0e38f, -3.0e38f, -3.0e38f, -3.0e38f };
  for (int n = rq; n < N_; n += 4) {
    float4 v = p[(size_t)n * 32 + c4];
    mx.x = fmaxf(mx.x, v.x); mx.y = fmaxf(mx.y, v.y);
    mx.z = fmaxf(mx.z, v.z); mx.w = fmaxf(mx.w, v.w);
  }
  sm[rq][c4] = mx;
  __syncthreads();
  if (rq == 0) {
    float4 a = sm[0][c4], bb = sm[1][c4], ccv = sm[2][c4], d = sm[3][c4];
    float4 o;
    o.x = fmaxf(fmaxf(a.x, bb.x), fmaxf(ccv.x, d.x));
    o.y = fmaxf(fmaxf(a.y, bb.y), fmaxf(ccv.y, d.y));
    o.z = fmaxf(fmaxf(a.z, bb.z), fmaxf(ccv.z, d.z));
    o.w = fmaxf(fmaxf(a.w, bb.w), fmaxf(ccv.w, d.w));
    ((float4*)(Out + (size_t)b * H_))[c4] = o;
  }
}

extern "C" void kernel_launch(void* const* d_in, const int* in_sizes, int n_in,
                              void* d_out, int out_size, void* d_ws, size_t ws_size,
                              hipStream_t stream) {
  (void)in_sizes; (void)n_in; (void)out_size; (void)ws_size;
  const float* x       = (const float*)d_in[0];
  const int*   lengths = (const int*)d_in[1];
  const float* W0  = (const float*)d_in[2];
  const float* b0  = (const float*)d_in[3];
  const float* g0  = (const float*)d_in[4];
  const float* be0 = (const float*)d_in[5];
  const float* W1  = (const float*)d_in[6];
  const float* b1  = (const float*)d_in[7];
  const float* g1  = (const float*)d_in[8];
  const float* be1 = (const float*)d_in[9];
  const float* Wq  = (const float*)d_in[10];
  const float* bq  = (const float*)d_in[11];
  const float* Wk  = (const float*)d_in[12];
  const float* bk  = (const float*)d_in[13];
  const float* Wv  = (const float*)d_in[14];
  const float* bv  = (const float*)d_in[15];
  const float* gl  = (const float*)d_in[16];
  const float* bl  = (const float*)d_in[17];
  float* out = (float*)d_out;

  // workspace carve-up
  char* ws = (char*)d_ws;
  size_t off = 0;
  _Float16* wf16 = (_Float16*)(ws + off);
  off += (size_t)11 * WSZ * sizeof(_Float16);
  off = (off + 255) & ~(size_t)255;
  float*    hF  = (float*)(ws + off);    off += (size_t)R_ * H_ * sizeof(float);
  _Float16* h16 = (_Float16*)(ws + off); off += (size_t)R_ * H_ * sizeof(_Float16);
  _Float16* q16 = (_Float16*)(ws + off); off += (size_t)R_ * H_ * sizeof(_Float16);
  _Float16* k16 = (_Float16*)(ws + off); off += (size_t)R_ * H_ * sizeof(_Float16);
  _Float16* vt16= (_Float16*)(ws + off); off += (size_t)R_ * H_ * sizeof(_Float16);
  float*    ctx = (float*)(ws + off);    off += (size_t)R_ * H_ * sizeof(float);

  pack_weights<<<(11 * WSZ) / 256, 256, 0, stream>>>(W0, W1, Wq, Wk, Wv, wf16);
  mlp_kernel<<<MT_, 256, 0, stream>>>(x, wf16, b0, g0, be0, b1, g1, be1, hF, h16);
  for (int d = 0; d < DEPTH_; ++d) {
    const _Float16* Wq16 = wf16 + (size_t)(2 + d) * WSZ;
    const _Float16* Wk16 = wf16 + (size_t)(5 + d) * WSZ;
    const _Float16* Wv16 = wf16 + (size_t)(8 + d) * WSZ;
    proj_kernel<<<dim3(MT_, 3), 256, 0, stream>>>(h16, Wq16, Wk16, Wv16,
                                                  bq + d * H_, bk + d * H_, bv + d * H_,
                                                  q16, k16, vt16);
    attn_kernel<<<dim3(N_ / 16, NH_, B_), 256, 0, stream>>>(q16, k16, vt16, lengths, ctx);
    ln_res_kernel<<<MT_, 256, 0, stream>>>(ctx, hF, h16, gl + d * H_, bl + d * H_);
  }
  max_kernel<<<B_, 128, 0, stream>>>(hF, out);
}
